// EncoderTransformer_82892868813529
// MI455X (gfx1250) — matrix operand order repaired and checked by reading, not compiled, not run on hardware
//
#include <hip/hip_runtime.h>
#include <hip/hip_bf16.h>
#include <math.h>

// ---------------------------------------------------------------------------
// EncoderTransformer (cross-attn + BiMamba + FFN, 2 layers) for MI455X gfx1250
//   - All dense GEMMs + attention through v_wmma_f32_16x16x32_f16 (wave32)
//   - f16 weight/activation caching: GEMM staging is pure 128-bit copies
//   - Double-buffered LDS tiles (global loads overlap WMMA)
//   - Mamba 65536-step recurrence via 3-pass chunked associative scan
//   - Workspace requirement: ~62 MB
// ---------------------------------------------------------------------------

#define DMODEL 512
#define NHEADS 8
#define DHEAD  64
#define NLAY   2
#define NQ     128
#define BATCH  8
#define SKV    256          // concat([src_addition, src_addition])
#define NTOK   1024         // NQ * BATCH
#define KVTOK  2048         // SKV * BATCH
#define LSEQ   65536        // mamba sequence length (NQ * DMODEL)
#define DIN    2
#define DST    16
#define FFHID  2048
#define CHK    256          // scan chunk length
#define NCHK   256          // LSEQ / CHK

typedef __attribute__((ext_vector_type(16))) _Float16 v16h;
typedef __attribute__((ext_vector_type(8)))  _Float16 v8h;
typedef __attribute__((ext_vector_type(4)))  _Float16 v4h;
typedef __attribute__((ext_vector_type(8)))  float    v8f;

// ---- WMMA fragment loaders (contiguous k order, same in A and B, wave32) -------------------
// A 16x32 f16: lane<16 -> M=lane, elems0..15=K0..15
//              lane>=16 -> M=lane-16, elems0..15=K16..31
__device__ __forceinline__ v16h ld_frag_a(const _Float16* row, int lane) {
  const int kb = (lane >> 4) * 16;
  v16h a;
  ((v8h*)&a)[0] = *(const v8h*)(row + kb);
  ((v8h*)&a)[1] = *(const v8h*)(row + kb + 8);
  return a;
}
// B 32x16 f16 (stored [n][k] in LDS): lane half selects K 0..15 / 16..31;
// caller passes &tile[n*stride + (lane>>4)*16]
__device__ __forceinline__ v16h ld_frag_b(const _Float16* p) {
  v16h b;
  ((v8h*)&b)[0] = *(const v8h*)(p);
  ((v8h*)&b)[1] = *(const v8h*)(p + 8);
  return b;
}

__device__ __forceinline__ float siluf(float v)     { return v * (1.f / (1.f + __expf(-v))); }
__device__ __forceinline__ float softplusf(float v) { return v > 20.f ? v : log1pf(__expf(v)); }

// ---------------------------------------------------------------------------
// GEMM: C[M,N] = act(A[M,K] @ W[N,K]^T + bias[N]);  A,W f16; M,N %64, K %32.
// 64x64 block tile, 4 waves (2x2), wave = 2x2 WMMA tiles; double-buffered LDS.
// flags: bit0 = relu, bit1 = write f16 to Ch (else f32 to Cf)
// ---------------------------------------------------------------------------
#define GTS 40  // LDS row stride in halves (80 B rows: 16B-aligned, bank-skewed)

__global__ __launch_bounds__(128)
void gemm_wmma_kernel(const _Float16* __restrict__ A, int lda,
                      const _Float16* __restrict__ W, int ldw,
                      const float* __restrict__ bias,
                      float* __restrict__ Cf, _Float16* __restrict__ Ch, int ldc,
                      int M, int N, int K, int flags)
{
  __shared__ __align__(16) _Float16 As[2][64 * GTS];
  __shared__ __align__(16) _Float16 Ws[2][64 * GTS];
  const int tid = threadIdx.x;
  const int lane = tid & 31, wid = tid >> 5;
  const int waveN = wid & 1, waveM = wid >> 1;
  const int m0 = blockIdx.y * 64, n0 = blockIdx.x * 64;

  v8f acc[2][2];
#pragma unroll
  for (int i = 0; i < 2; ++i)
#pragma unroll
    for (int j = 0; j < 2; ++j) { v8f z = {}; acc[i][j] = z; }

  // per-thread staging registers: 2x 128-bit for A, 2x for W (tile = 64x32 h)
  v8h ra[2], rw[2];
  const int r0 = (tid + 0) >> 2,   c0 = ((tid + 0) & 3) * 8;
  const int r1 = (tid + 128) >> 2, c1 = ((tid + 128) & 3) * 8;

  auto load_tile = [&](int kk) {
    ra[0] = *(const v8h*)&A[(m0 + r0) * lda + kk + c0];
    ra[1] = *(const v8h*)&A[(m0 + r1) * lda + kk + c1];
    rw[0] = *(const v8h*)&W[(n0 + r0) * ldw + kk + c0];
    rw[1] = *(const v8h*)&W[(n0 + r1) * ldw + kk + c1];
  };
  auto stash_tile = [&](int buf) {
    *(v8h*)&As[buf][r0 * GTS + c0] = ra[0];
    *(v8h*)&As[buf][r1 * GTS + c1] = ra[1];
    *(v8h*)&Ws[buf][r0 * GTS + c0] = rw[0];
    *(v8h*)&Ws[buf][r1 * GTS + c1] = rw[1];
  };

  load_tile(0);
  stash_tile(0);
  int cur = 0;
  for (int kk = 0; kk < K; kk += 32) {
    __syncthreads();
    const bool nxt = (kk + 32) < K;
    if (nxt) load_tile(kk + 32);                 // overlap with WMMA below
    if (kk + 64 < K)                             // gfx1250 global_prefetch_b8
      __builtin_prefetch(&A[(m0 + r0) * lda + kk + 64], 0, 1);

    v16h af[2], bf[2];
#pragma unroll
    for (int i = 0; i < 2; ++i)
      af[i] = ld_frag_a(&As[cur][(waveM * 32 + 16 * i + (lane & 15)) * GTS], lane);
#pragma unroll
    for (int j = 0; j < 2; ++j)
      bf[j] = ld_frag_b(&Ws[cur][(waveN * 32 + 16 * j + (lane & 15)) * GTS + ((lane >> 4) * 16)]);
#pragma unroll
    for (int i = 0; i < 2; ++i)
#pragma unroll
      for (int j = 0; j < 2; ++j)
        acc[i][j] = __builtin_amdgcn_wmma_f32_16x16x32_f16(
            false, af[i], false, bf[j], (short)0, acc[i][j], false, false);

    if (nxt) stash_tile(cur ^ 1);
    cur ^= 1;
  }
  // epilogue: C/D layout -> m = 8*(lane>=16)+vgpr, n = lane&15
#pragma unroll
  for (int i = 0; i < 2; ++i)
#pragma unroll
    for (int j = 0; j < 2; ++j)
#pragma unroll
      for (int r = 0; r < 8; ++r) {
        const int m = m0 + waveM * 32 + 16 * i + (lane >> 4) * 8 + r;
        const int n = n0 + waveN * 32 + 16 * j + (lane & 15);
        float v = acc[i][j][r] + bias[n];
        if (flags & 1) v = fmaxf(v, 0.f);
        if (flags & 2) Ch[m * ldc + n] = (_Float16)v;
        else           Cf[m * ldc + n] = v;
      }
}

// ---------------------------------------------------------------------------
// Attention: one block per (row-tile of 32, head, batch); 4 waves; f16 in/out.
// Scores live in WMMA accumulators (wave owns 64 of 256 cols); softmax
// cross-wave via LDS; P->f16 LDS; V restaged transposed for second WMMA.
// ---------------------------------------------------------------------------
#define KSTR 72    // 144 B rows
#define PSTR 272   // 544 B rows
#define VSTR 264   // 528 B rows

__global__ __launch_bounds__(128)
void attn_kernel(const _Float16* __restrict__ qp, const _Float16* __restrict__ kvp,
                 _Float16* __restrict__ o)
{
  __shared__ __align__(16) _Float16 KV[SKV * KSTR];   // K tile, then V^T tile
  __shared__ __align__(16) _Float16 Qs[32 * KSTR];
  __shared__ __align__(16) _Float16 Ps[32 * PSTR];
  __shared__ float redm[32 * 4];
  __shared__ float reds[32 * 4];

  const int rt = blockIdx.x, hh = blockIdx.y, b = blockIdx.z;
  const int tid = threadIdx.x, lane = tid & 31, wave = tid >> 5;
  const int n0 = rt * 32;
  const int hcol = hh * DHEAD;

  // stage Q (32x64) and K (256x64) via 128-bit copies
#pragma unroll
  for (int i = 0; i < 2; ++i) {
    const int idx8 = tid + i * 128;
    const int r = idx8 >> 3, c8 = (idx8 & 7) * 8;
    *(v8h*)&Qs[r * KSTR + c8] =
        *(const v8h*)&qp[((n0 + r) * BATCH + b) * DMODEL + hcol + c8];
  }
#pragma unroll
  for (int i = 0; i < 16; ++i) {
    const int idx8 = tid + i * 128;
    const int s = idx8 >> 3, c8 = (idx8 & 7) * 8;
    *(v8h*)&KV[s * KSTR + c8] =
        *(const v8h*)&kvp[(s * BATCH + b) * (2 * DMODEL) + hcol + c8];
  }
  __syncthreads();

  // scores: 32 rows x (wave-owned 64 cols) = 2x4 WMMA tiles
  v8f acc[2][4];
#pragma unroll
  for (int i = 0; i < 2; ++i)
#pragma unroll
    for (int j = 0; j < 4; ++j) { v8f z = {}; acc[i][j] = z; }

#pragma unroll
  for (int kt = 0; kt < DHEAD; kt += 32) {
    v16h af[2];
#pragma unroll
    for (int i = 0; i < 2; ++i)
      af[i] = ld_frag_a(&Qs[(16 * i + (lane & 15)) * KSTR + kt], lane);
#pragma unroll
    for (int j = 0; j < 4; ++j) {
      const _Float16* pb =
          &KV[(wave * 64 + 16 * j + (lane & 15)) * KSTR + kt + ((lane >> 4) * 16)];
      v16h bf = ld_frag_b(pb);
#pragma unroll
      for (int i = 0; i < 2; ++i)
        acc[i][j] = __builtin_amdgcn_wmma_f32_16x16x32_f16(
            false, af[i], false, bf, (short)0, acc[i][j], false, false);
    }
  }
  const float scale = 0.125f;  // 1/sqrt(64)
#pragma unroll
  for (int i = 0; i < 2; ++i)
#pragma unroll
    for (int j = 0; j < 4; ++j)
#pragma unroll
      for (int r = 0; r < 8; ++r) acc[i][j][r] *= scale;

  // row max: in-register over j, butterfly over 16 cols, LDS across waves
  float rmax[2][8];
#pragma unroll
  for (int i = 0; i < 2; ++i)
#pragma unroll
    for (int r = 0; r < 8; ++r) {
      float m = acc[i][0][r];
#pragma unroll
      for (int j = 1; j < 4; ++j) m = fmaxf(m, acc[i][j][r]);
#pragma unroll
      for (int t = 8; t; t >>= 1) m = fmaxf(m, __shfl_xor(m, t, 16));
      if ((lane & 15) == 0) redm[(16 * i + (lane >> 4) * 8 + r) * 4 + wave] = m;
    }
  __syncthreads();  // also fences all KV(K) reads
#pragma unroll
  for (int i = 0; i < 2; ++i)
#pragma unroll
    for (int r = 0; r < 8; ++r) {
      const int row = 16 * i + (lane >> 4) * 8 + r;
      rmax[i][r] = fmaxf(fmaxf(redm[row * 4 + 0], redm[row * 4 + 1]),
                         fmaxf(redm[row * 4 + 2], redm[row * 4 + 3]));
    }
  // exp in place + row sums
#pragma unroll
  for (int i = 0; i < 2; ++i)
#pragma unroll
    for (int r = 0; r < 8; ++r) {
      float ssum = 0.f;
#pragma unroll
      for (int j = 0; j < 4; ++j) {
        const float e = __expf(acc[i][j][r] - rmax[i][r]);
        acc[i][j][r] = e;
        ssum += e;
      }
#pragma unroll
      for (int t = 8; t; t >>= 1) ssum += __shfl_xor(ssum, t, 16);
      if ((lane & 15) == 0) reds[(16 * i + (lane >> 4) * 8 + r) * 4 + wave] = ssum;
    }
  // restage V transposed into KV (K reads fenced by previous barrier):
  // 128-bit global loads, scalar b16 LDS stores (inherent to transpose)
#pragma unroll
  for (int i = 0; i < 16; ++i) {
    const int idx8 = tid + i * 128;
    const int s = idx8 >> 3, c8 = (idx8 & 7) * 8;
    const v8h vv =
        *(const v8h*)&kvp[(s * BATCH + b) * (2 * DMODEL) + DMODEL + hcol + c8];
#pragma unroll
    for (int q = 0; q < 8; ++q) KV[(c8 + q) * VSTR + s] = vv[q];
  }
  __syncthreads();
  // finalize P as f16
#pragma unroll
  for (int i = 0; i < 2; ++i)
#pragma unroll
    for (int r = 0; r < 8; ++r) {
      const int row = 16 * i + (lane >> 4) * 8 + r;
      const float rinv = 1.f / (reds[row * 4 + 0] + reds[row * 4 + 1] +
                                reds[row * 4 + 2] + reds[row * 4 + 3]);
#pragma unroll
      for (int j = 0; j < 4; ++j) {
        const int col = wave * 64 + 16 * j + (lane & 15);
        Ps[row * PSTR + col] = (_Float16)(acc[i][j][r] * rinv);
      }
    }
  __syncthreads();

  // O = P(32x256) @ V(256x64); each wave owns 16 output cols
  v8f oacc[2];
  { v8f z = {}; oacc[0] = z; oacc[1] = z; }
#pragma unroll
  for (int st = 0; st < 8; ++st) {
    const int s0 = st * 32;
    const _Float16* pb = &KV[(wave * 16 + (lane & 15)) * VSTR + s0 + ((lane >> 4) * 16)];
    v16h bf = ld_frag_b(pb);
#pragma unroll
    for (int i = 0; i < 2; ++i) {
      v16h af = ld_frag_a(&Ps[(16 * i + (lane & 15)) * PSTR + s0], lane);
      oacc[i] = __builtin_amdgcn_wmma_f32_16x16x32_f16(
          false, af, false, bf, (short)0, oacc[i], false, false);
    }
  }
#pragma unroll
  for (int i = 0; i < 2; ++i)
#pragma unroll
    for (int r = 0; r < 8; ++r) {
      const int n = n0 + 16 * i + (lane >> 4) * 8 + r;
      const int c = hcol + wave * 16 + (lane & 15);
      o[(n * BATCH + b) * DMODEL + c] = (_Float16)oacc[i][r];
    }
}

// ---------------------------------------------------------------------------
// Residual-add + LayerNorm over D=512; also emits f16 copy of x for GEMMs.
// ---------------------------------------------------------------------------
__global__ __launch_bounds__(128)
void add_ln_kernel(float* __restrict__ x, const float* __restrict__ res,
                   const float* __restrict__ w, const float* __restrict__ bb,
                   _Float16* __restrict__ xh)
{
  const int row = blockIdx.x, tid = threadIdx.x;
  __shared__ float sred[4];
  float v[4];
  float s = 0.f;
#pragma unroll
  for (int i = 0; i < 4; ++i) {
    const int c = tid + i * 128;
    v[i] = x[row * DMODEL + c] + res[row * DMODEL + c];
    s += v[i];
  }
#pragma unroll
  for (int t = 16; t; t >>= 1) s += __shfl_xor(s, t, 32);
  if ((tid & 31) == 0) sred[tid >> 5] = s;
  __syncthreads();
  const float mu = (sred[0] + sred[1] + sred[2] + sred[3]) * (1.f / 512.f);
  float q = 0.f;
#pragma unroll
  for (int i = 0; i < 4; ++i) { const float d = v[i] - mu; q += d * d; }
#pragma unroll
  for (int t = 16; t; t >>= 1) q += __shfl_xor(q, t, 32);
  __syncthreads();
  if ((tid & 31) == 0) sred[tid >> 5] = q;
  __syncthreads();
  const float var = (sred[0] + sred[1] + sred[2] + sred[3]) * (1.f / 512.f);
  const float rstd = rsqrtf(var + 1e-5f);
#pragma unroll
  for (int i = 0; i < 4; ++i) {
    const int c = tid + i * 128;
    const float o = (v[i] - mu) * rstd * w[c] + bb[c];
    x[row * DMODEL + c] = o;
    xh[row * DMODEL + c] = (_Float16)o;
  }
}

// ---------------------------------------------------------------------------
// f32 -> f16 vectorized conversion (n multiple of 4)
// ---------------------------------------------------------------------------
__global__ void cvt_f16_kernel(const float* __restrict__ in,
                               _Float16* __restrict__ out, int n4)
{
  const int idx = blockIdx.x * blockDim.x + threadIdx.x;
  if (idx >= n4) return;
  const float4 v = ((const float4*)in)[idx];
  v4h h;
  h[0] = (_Float16)v.x; h[1] = (_Float16)v.y;
  h[2] = (_Float16)v.z; h[3] = (_Float16)v.w;
  ((v4h*)out)[idx] = h;
}

// build f16 ka = concat([src_addition, src_addition]) (vectorized by 4)
__global__ void build_ka_kernel(const float* __restrict__ sa,
                                _Float16* __restrict__ kah)
{
  const int idx = blockIdx.x * blockDim.x + threadIdx.x;
  const int half4 = NQ * BATCH * DMODEL / 4;
  if (idx >= 2 * half4) return;
  const float4 v = ((const float4*)sa)[idx >= half4 ? idx - half4 : idx];
  v4h h;
  h[0] = (_Float16)v.x; h[1] = (_Float16)v.y;
  h[2] = (_Float16)v.z; h[3] = (_Float16)v.w;
  ((v4h*)kah)[idx] = h;
}

// ---------------------------------------------------------------------------
// Mamba elementwise / scan kernels (f32 state; 3-pass chunked scan)
// ---------------------------------------------------------------------------
__global__ void mamba_prep_kernel(const float* __restrict__ u,
                                  const float* __restrict__ inw,
                                  float* __restrict__ x2, float* __restrict__ z)
{
  const int idx = blockIdx.x * blockDim.x + threadIdx.x;
  if (idx >= BATCH * LSEQ) return;
  const float uv = u[idx];
  x2[idx * 2 + 0] = uv * inw[0];
  x2[idx * 2 + 1] = uv * inw[1];
  z[idx * 2 + 0]  = uv * inw[2];
  z[idx * 2 + 1]  = uv * inw[3];
}

__global__ void mamba_conv_kernel(const float* __restrict__ x2,
                                  const float* __restrict__ cw,
                                  const float* __restrict__ cb,
                                  float* __restrict__ xc, int dir)
{
  const int idx = blockIdx.x * blockDim.x + threadIdx.x;
  if (idx >= BATCH * LSEQ * DIN) return;
  const int d = idx & 1;
  const int bl = idx >> 1;
  const int l = bl & (LSEQ - 1);
  const int b = bl >> 16;
  float acc = cb[d];
#pragma unroll
  for (int k = 0; k < 4; ++k) {
    const int lp = l - 3 + k;
    if (lp >= 0) {
      const int lsrc = dir ? (LSEQ - 1 - lp) : lp;
      acc += x2[(b * LSEQ + lsrc) * 2 + d] * cw[d * 4 + k];
    }
  }
  xc[(b * LSEQ + l) * 2 + d] = siluf(acc);
}

// pass 1: per-chunk (prod a, acc b) with h0 = 0.  32 lanes = (chan d, state s).
__global__ __launch_bounds__(32)
void mamba_pass1_kernel(const float* __restrict__ xcf, const float* __restrict__ xcb,
                        const float* __restrict__ xwf, const float* __restrict__ xwb,
                        const float* __restrict__ dtwf, const float* __restrict__ dtwb,
                        const float* __restrict__ dtbf, const float* __restrict__ dtbb,
                        const float* __restrict__ Alf, const float* __restrict__ Alb,
                        float* __restrict__ Aprod, float* __restrict__ Bacc)
{
  const int ch = blockIdx.x, dir = blockIdx.y, b = blockIdx.z;
  const int lane = threadIdx.x, d = lane >> 4, s = lane & 15;
  const float* xc = dir ? xcb : xcf;
  const float* xw = dir ? xwb : xwf;
  const float dtw = (dir ? dtwb : dtwf)[d];
  const float dtb = (dir ? dtbb : dtbf)[d];
  const float Ads = -__expf((dir ? Alb : Alf)[d * 16 + s]);
  const float dt0 = xw[0], dt1 = xw[1];
  const float Bw0 = xw[(1 + s) * 2 + 0], Bw1 = xw[(1 + s) * 2 + 1];
  float ap = 1.f, bc = 0.f;
  const int base = (b * LSEQ + ch * CHK) * 2;
  for (int t = 0; t < CHK; ++t) {
    const float xc0 = xc[base + t * 2 + 0];
    const float xc1 = xc[base + t * 2 + 1];
    const float delta = softplusf((dt0 * xc0 + dt1 * xc1) * dtw + dtb);
    const float a = __expf(delta * Ads);
    const float bu = delta * (Bw0 * xc0 + Bw1 * xc1) * (d ? xc1 : xc0);
    ap *= a;
    bc = a * bc + bu;
  }
  const int o = ((b * 2 + dir) * NCHK + ch) * 32 + lane;
  Aprod[o] = ap;
  Bacc[o] = bc;
}

// pass 2: 256-step chunk-level recurrence, emits per-chunk initial states
__global__ __launch_bounds__(32)
void mamba_pass2_kernel(const float* __restrict__ Aprod, const float* __restrict__ Bacc,
                        float* __restrict__ Hinit)
{
  const int dir = blockIdx.x, b = blockIdx.y, lane = threadIdx.x;
  float h = 0.f;
  const int base = (b * 2 + dir) * NCHK;
  for (int ch = 0; ch < NCHK; ++ch) {
    const int o = (base + ch) * 32 + lane;
    Hinit[o] = h;
    h = Aprod[o] * h + Bacc[o];
  }
}

// pass 3: replay chunk with correct h0, produce y (state reduce via shfl_xor)
__global__ __launch_bounds__(32)
void mamba_pass3_kernel(const float* __restrict__ xcf, const float* __restrict__ xcb,
                        const float* __restrict__ xwf, const float* __restrict__ xwb,
                        const float* __restrict__ dtwf, const float* __restrict__ dtwb,
                        const float* __restrict__ dtbf, const float* __restrict__ dtbb,
                        const float* __restrict__ Alf, const float* __restrict__ Alb,
                        const float* __restrict__ Df, const float* __restrict__ Db,
                        const float* __restrict__ Hinit, float* __restrict__ yout)
{
  const int ch = blockIdx.x, dir = blockIdx.y, b = blockIdx.z;
  const int lane = threadIdx.x, d = lane >> 4, s = lane & 15;
  const float* xc = dir ? xcb : xcf;
  const float* xw = dir ? xwb : xwf;
  const float dtw = (dir ? dtwb : dtwf)[d];
  const float dtb = (dir ? dtbb : dtbf)[d];
  const float Ads = -__expf((dir ? Alb : Alf)[d * 16 + s]);
  const float Dd = (dir ? Db : Df)[d];
  const float dt0 = xw[0], dt1 = xw[1];
  const float Bw0 = xw[(1 + s) * 2 + 0],  Bw1 = xw[(1 + s) * 2 + 1];
  const float Cw0 = xw[(17 + s) * 2 + 0], Cw1 = xw[(17 + s) * 2 + 1];
  float h = Hinit[((b * 2 + dir) * NCHK + ch) * 32 + lane];
  const int base = (b * LSEQ + ch * CHK) * 2;
  for (int t = 0; t < CHK; ++t) {
    const float xc0 = xc[base + t * 2 + 0];
    const float xc1 = xc[base + t * 2 + 1];
    const float xcd = d ? xc1 : xc0;
    const float delta = softplusf((dt0 * xc0 + dt1 * xc1) * dtw + dtb);
    const float a = __expf(delta * Ads);
    const float bu = delta * (Bw0 * xc0 + Bw1 * xc1) * xcd;
    h = a * h + bu;
    float yv = h * (Cw0 * xc0 + Cw1 * xc1);
#pragma unroll
    for (int m = 8; m; m >>= 1) yv += __shfl_xor(yv, m, 16);
    if (s == 0)
      yout[((dir * BATCH + b) * LSEQ + ch * CHK + t) * 2 + d] = yv + Dd * xcd;
  }
}

__global__ void mamba_combine_kernel(const float* __restrict__ yboth,
                                     const float* __restrict__ z,
                                     const float* __restrict__ ow,
                                     float* __restrict__ uout)
{
  const int idx = blockIdx.x * blockDim.x + threadIdx.x;
  if (idx >= BATCH * LSEQ) return;
  const int b = idx >> 16, l = idx & (LSEQ - 1);
  float acc = 0.f;
#pragma unroll
  for (int d = 0; d < 2; ++d) {
    const float yf = yboth[((0 * BATCH + b) * LSEQ + l) * 2 + d];
    const float yb = yboth[((1 * BATCH + b) * LSEQ + (LSEQ - 1 - l)) * 2 + d];
    acc += (yf + yb) * siluf(z[idx * 2 + d]) * ow[d];
  }
  uout[idx] = acc;
}

// ---------------------------------------------------------------------------
__global__ void copy_kernel(const float* __restrict__ a, float* __restrict__ b, int n)
{
  const int idx = blockIdx.x * blockDim.x + threadIdx.x;
  if (idx < n) b[idx] = a[idx];
}

// ---------------------------------------------------------------------------
extern "C" void kernel_launch(void* const* d_in, const int* in_sizes, int n_in,
                              void* d_out, int out_size, void* d_ws, size_t ws_size,
                              hipStream_t stream)
{
  const float* src      = (const float*)d_in[0];
  const float* srca     = (const float*)d_in[1];
  const float* ca_in_w  = (const float*)d_in[2];
  const float* ca_in_b  = (const float*)d_in[3];
  const float* ca_out_w = (const float*)d_in[4];
  const float* ca_out_b = (const float*)d_in[5];
  const float* m_in_w   = (const float*)d_in[6];
  const float* m_cw     = (const float*)d_in[7];
  const float* m_cb     = (const float*)d_in[8];
  const float* m_cw_b   = (const float*)d_in[9];
  const float* m_cb_b   = (const float*)d_in[10];
  const float* m_xw     = (const float*)d_in[11];
  const float* m_xw_b   = (const float*)d_in[12];
  const float* m_dtw    = (const float*)d_in[13];
  const float* m_dtb    = (const float*)d_in[14];
  const float* m_dtw_b  = (const float*)d_in[15];
  const float* m_dtb_b  = (const float*)d_in[16];
  const float* m_Al     = (const float*)d_in[17];
  const float* m_Al_b   = (const float*)d_in[18];
  const float* m_D      = (const float*)d_in[19];
  const float* m_D_b    = (const float*)d_in[20];
  const float* m_ow     = (const float*)d_in[21];
  const float* ff1_w    = (const float*)d_in[22];
  const float* ff1_b    = (const float*)d_in[23];
  const float* ff2_w    = (const float*)d_in[24];
  const float* ff2_b    = (const float*)d_in[25];
  const float* ln1_w    = (const float*)d_in[26];
  const float* ln1_b    = (const float*)d_in[27];
  const float* ln2_w    = (const float*)d_in[28];
  const float* ln2_b    = (const float*)d_in[29];
  (void)in_sizes; (void)n_in; (void)ws_size;

  // ---- workspace carve-out: f32 region then f16 region (~62 MB total) ----
  float* p = (float*)d_ws;
  float* xbuf  = p; p += NTOK * DMODEL;          // 524288
  float* s2    = p; p += NTOK * DMODEL;          // 524288
  float* x2    = p; p += BATCH * LSEQ * 2;       // 1048576
  float* zb    = p; p += BATCH * LSEQ * 2;       // 1048576
  float* xcf   = p; p += BATCH * LSEQ * 2;       // 1048576
  float* xcb   = p; p += BATCH * LSEQ * 2;       // 1048576
  float* Ap    = p; p += BATCH * 2 * NCHK * 32;  // 131072
  float* Bc    = p; p += BATCH * 2 * NCHK * 32;  // 131072
  float* Hin   = p; p += BATCH * 2 * NCHK * 32;  // 131072
  float* yb2   = p; p += 2 * BATCH * LSEQ * 2;   // 2097152
  float* uout  = p; p += BATCH * LSEQ;           // 524288
  float* ffout = p; p += NTOK * DMODEL;          // 524288
  _Float16* hp = (_Float16*)p;
  _Float16* xh     = hp; hp += NTOK * DMODEL;         // activations f16
  _Float16* kah    = hp; hp += KVTOK * DMODEL;
  _Float16* qph    = hp; hp += NTOK * DMODEL;
  _Float16* kvph   = hp; hp += KVTOK * 2 * DMODEL;
  _Float16* attnoh = hp; hp += NTOK * DMODEL;
  _Float16* hidh   = hp; hp += NTOK * FFHID;
  _Float16* caw_h  = hp; hp += NLAY * 3 * DMODEL * DMODEL;  // weights f16
  _Float16* cao_h  = hp; hp += NLAY * DMODEL * DMODEL;
  _Float16* ff1_h  = hp; hp += NLAY * FFHID * DMODEL;
  _Float16* ff2_h  = hp; hp += NLAY * DMODEL * FFHID;

  const int NE = NTOK * DMODEL;  // 524288 = BATCH*LSEQ

  // one-time f16 weight caching + inputs
  copy_kernel<<<(NE + 255) / 256, 256, 0, stream>>>(src, xbuf, NE);
  cvt_f16_kernel<<<(NE / 4 + 255) / 256, 256, 0, stream>>>(src, xh, NE / 4);
  build_ka_kernel<<<(2 * NE / 4 + 255) / 256, 256, 0, stream>>>(srca, kah);
  {
    const int n1 = NLAY * 3 * DMODEL * DMODEL / 4;
    cvt_f16_kernel<<<(n1 + 255) / 256, 256, 0, stream>>>(ca_in_w, caw_h, n1);
    const int n2 = NLAY * DMODEL * DMODEL / 4;
    cvt_f16_kernel<<<(n2 + 255) / 256, 256, 0, stream>>>(ca_out_w, cao_h, n2);
    const int n3 = NLAY * FFHID * DMODEL / 4;
    cvt_f16_kernel<<<(n3 + 255) / 256, 256, 0, stream>>>(ff1_w, ff1_h, n3);
    cvt_f16_kernel<<<(n3 + 255) / 256, 256, 0, stream>>>(ff2_w, ff2_h, n3);
  }

  for (int L = 0; L < NLAY; ++L) {
    const _Float16* inwh = caw_h + (size_t)L * 3 * DMODEL * DMODEL;
    const float*    inb  = ca_in_b + (size_t)L * 3 * DMODEL;
    // Q projection (f16 out): (1024x512) = x @ Wq^T
    gemm_wmma_kernel<<<dim3(DMODEL / 64, NTOK / 64), 128, 0, stream>>>(
        xh, DMODEL, inwh, DMODEL, inb, nullptr, qph, DMODEL,
        NTOK, DMODEL, DMODEL, 2);
    // fused K|V projection (f16 out): (2048x1024) = ka @ [Wk;Wv]^T
    gemm_wmma_kernel<<<dim3(2 * DMODEL / 64, KVTOK / 64), 128, 0, stream>>>(
        kah, DMODEL, inwh + (size_t)DMODEL * DMODEL, DMODEL, inb + DMODEL,
        nullptr, kvph, 2 * DMODEL, KVTOK, 2 * DMODEL, DMODEL, 2);
    attn_kernel<<<dim3(NQ / 32, NHEADS, BATCH), 128, 0, stream>>>(qph, kvph, attnoh);
    // out projection (f32 out, feeds mamba + residual)
    gemm_wmma_kernel<<<dim3(DMODEL / 64, NTOK / 64), 128, 0, stream>>>(
        attnoh, DMODEL, cao_h + (size_t)L * DMODEL * DMODEL, DMODEL,
        ca_out_b + (size_t)L * DMODEL, s2, nullptr, DMODEL,
        NTOK, DMODEL, DMODEL, 0);

    // --- BiMamba on u = s2 reinterpreted as (8, 65536) ---
    mamba_prep_kernel<<<(NE + 255) / 256, 256, 0, stream>>>(s2, m_in_w + L * 4, x2, zb);
    mamba_conv_kernel<<<(2 * NE + 255) / 256, 256, 0, stream>>>(
        x2, m_cw + L * 8, m_cb + L * 2, xcf, 0);
    mamba_conv_kernel<<<(2 * NE + 255) / 256, 256, 0, stream>>>(
        x2, m_cw_b + L * 8, m_cb_b + L * 2, xcb, 1);
    mamba_pass1_kernel<<<dim3(NCHK, 2, BATCH), 32, 0, stream>>>(
        xcf, xcb, m_xw + L * 66, m_xw_b + L * 66, m_dtw + L * 2, m_dtw_b + L * 2,
        m_dtb + L * 2, m_dtb_b + L * 2, m_Al + L * 32, m_Al_b + L * 32, Ap, Bc);
    mamba_pass2_kernel<<<dim3(2, BATCH), 32, 0, stream>>>(Ap, Bc, Hin);
    mamba_pass3_kernel<<<dim3(NCHK, 2, BATCH), 32, 0, stream>>>(
        xcf, xcb, m_xw + L * 66, m_xw_b + L * 66, m_dtw + L * 2, m_dtw_b + L * 2,
        m_dtb + L * 2, m_dtb_b + L * 2, m_Al + L * 32, m_Al_b + L * 32,
        m_D + L * 2, m_D_b + L * 2, Hin, yb2);
    mamba_combine_kernel<<<(NE + 255) / 256, 256, 0, stream>>>(
        yb2, zb, m_ow + L * 2, uout);

    add_ln_kernel<<<NTOK, 128, 0, stream>>>(xbuf, uout,
        ln1_w + L * DMODEL, ln1_b + L * DMODEL, xh);

    // --- FFN ---
    gemm_wmma_kernel<<<dim3(FFHID / 64, NTOK / 64), 128, 0, stream>>>(
        xh, DMODEL, ff1_h + (size_t)L * FFHID * DMODEL, DMODEL,
        ff1_b + (size_t)L * FFHID, nullptr, hidh, FFHID,
        NTOK, FFHID, DMODEL, 1 | 2);
    gemm_wmma_kernel<<<dim3(DMODEL / 64, NTOK / 64), 128, 0, stream>>>(
        hidh, FFHID, ff2_h + (size_t)L * DMODEL * FFHID, FFHID,
        ff2_b + (size_t)L * DMODEL, ffout, nullptr, DMODEL,
        NTOK, DMODEL, FFHID, 0);
    add_ln_kernel<<<NTOK, 128, 0, stream>>>(xbuf, ffout,
        ln2_w + L * DMODEL, ln2_b + L * DMODEL, xh);
  }

  copy_kernel<<<(NE + 255) / 256, 256, 0, stream>>>(xbuf, (float*)d_out, NE);
}
